// Attention_5643587027315
// MI455X (gfx1250) — matrix operand order repaired and checked by reading, not compiled, not run on hardware
//
#include <hip/hip_runtime.h>

// ---------------------------------------------------------------------------
// Types for CDNA5 WMMA (wave32): bf16 16x16x32, f32 accumulate
// ---------------------------------------------------------------------------
typedef __bf16 bf16_t;
typedef __attribute__((ext_vector_type(16))) __bf16 v16bf;
typedef __attribute__((ext_vector_type(8)))  float  v8f;
typedef __attribute__((ext_vector_type(4)))  unsigned int u32x4;
typedef __attribute__((ext_vector_type(4)))  float  f32x4;

union ABf { v16bf v; u32x4 u[2]; };

#define WMMA_BF16(A, B, C) \
  __builtin_amdgcn_wmma_f32_16x16x32_bf16(false, (A), false, (B), (short)0, (C), false, false)

// Problem constants
constexpr int BB = 2, SS = 2048, DM = 1024, HH = 16, DH = 64;
constexpr float SCALE = 0.125f;  // 1/sqrt(64)
constexpr int LPAD = 72;         // LDS row stride (elems): 64 + 8 -> conflict-free b128 reads

__device__ __forceinline__ unsigned short f2bf(float f) {
  union { float f; unsigned int u; } t; t.f = f;
  unsigned int u = t.u;
  return (unsigned short)((u + 0x7FFFu + ((u >> 16) & 1u)) >> 16);  // RNE
}
__device__ __forceinline__ unsigned int pack2(float lo, float hi) {
  return (unsigned int)f2bf(lo) | ((unsigned int)f2bf(hi) << 16);
}
__device__ __forceinline__ v8f vzero8() {
  v8f z;
#pragma unroll
  for (int i = 0; i < 8; ++i) z[i] = 0.0f;
  return z;
}

// ---- CDNA5 async global->LDS copy (ASYNCcnt), via inline asm (VGLOBAL GV form:
//      vdst = LDS byte-address VGPR, vaddr = 64-bit global address, SADDR=off) ----
__device__ __forceinline__ void async_load_b128(const bf16_t* g, bf16_t* l) {
  unsigned lo = (unsigned)(size_t)l;  // LDS aperture: low 32 bits are the LDS offset
  asm volatile("global_load_async_to_lds_b128 %0, %1, off" :: "v"(lo), "v"(g) : "memory");
}
__device__ __forceinline__ void wait_async0() {
#if __has_builtin(__builtin_amdgcn_s_wait_asynccnt)
  __builtin_amdgcn_s_wait_asynccnt(0);
#else
  asm volatile("s_wait_asynccnt 0" ::: "memory");
#endif
}

// ---------------------------------------------------------------------------
// Kernel 0: fp32 -> bf16 conversion (x, qkv_w, out_w)
// ---------------------------------------------------------------------------
__global__ __launch_bounds__(256) void cvt_f32_to_bf16(const float* __restrict__ in,
                                                       unsigned short* __restrict__ out,
                                                       int n) {
  int i = blockIdx.x * blockDim.x + threadIdx.x;
  int stride = gridDim.x * blockDim.x;
  for (; i < n; i += stride) out[i] = f2bf(in[i]);
}

// ---------------------------------------------------------------------------
// Kernel 1: QKV projection + RoPE.
//   q/k: transposed orientation -> RoPE in-lane -> Q/K [B,H,S,64] bf16
//   v:   normal orientation -> stored transposed VT [B,H,64,S] bf16
// ---------------------------------------------------------------------------
__global__ __launch_bounds__(256) void qkv_rope_kernel(
    const bf16_t* __restrict__ xb,    // [4096,1024]
    const bf16_t* __restrict__ wb,    // [3072,1024]
    const float*  __restrict__ rope_freqs,  // [32]
    bf16_t* __restrict__ Qo, bf16_t* __restrict__ Ko, bf16_t* __restrict__ VTo) {
  const int lane = threadIdx.x & 31;
  const int wid  = blockIdx.x * 8 + (threadIdx.x >> 5);
  const int mt    = wid & 255;
  const int h     = (wid >> 8) & 15;
  const int which = wid >> 12;          // 0=q 1=k 2=v
  const int t0   = mt * 16;
  const int half = lane >> 4;
  const int l16  = lane & 15;

  v8f acc[4];
#pragma unroll
  for (int i = 0; i < 4; ++i) acc[i] = vzero8();

  if (which < 2) {
    const int nbase = which * 1024 + h * 64;
    for (int k0 = 0; k0 < DM; k0 += 32) {
      ABf Bm;  // B = x^T : lane = token
      const bf16_t* xr = xb + (size_t)(t0 + l16) * DM + k0 + half * 8;
      Bm.u[0] = *(const u32x4*)(xr);
      Bm.u[1] = *(const u32x4*)(xr + 16);
#pragma unroll
      for (int dt = 0; dt < 4; ++dt) {
        ABf Am;  // A = W rows
        const bf16_t* wr = wb + (size_t)(nbase + dt * 16 + l16) * DM + k0 + half * 8;
        Am.u[0] = *(const u32x4*)(wr);
        Am.u[1] = *(const u32x4*)(wr + 16);
        acc[dt] = WMMA_BF16(Am.v, Bm.v, acc[dt]);
      }
    }
    const int tok = t0 + l16;
    const int b   = tok >> 11;
    const int s   = tok & 2047;
    const float pos = (float)s;
#pragma unroll
    for (int dt = 0; dt < 2; ++dt) {
#pragma unroll
      for (int j = 0; j < 8; ++j) {
        const int i = dt * 16 + half * 8 + j;   // i == d in [0,32)
        float sn, cs;
        __sincosf(pos * rope_freqs[i], &sn, &cs);
        const float t1 = acc[dt][j];
        const float t2 = acc[dt + 2][j];
        acc[dt][j]     = t1 * cs - t2 * sn;
        acc[dt + 2][j] = t1 * sn + t2 * cs;
      }
    }
    const float sc = (which == 0) ? SCALE : 1.0f;
    bf16_t* dst = (which == 0 ? Qo : Ko) + ((size_t)(b * HH + h) * SS + s) * DH;
#pragma unroll
    for (int dt = 0; dt < 4; ++dt) {
      u32x4 pk;
#pragma unroll
      for (int jj = 0; jj < 4; ++jj)
        pk[jj] = pack2(acc[dt][2 * jj] * sc, acc[dt][2 * jj + 1] * sc);
      *(u32x4*)(dst + dt * 16 + half * 8) = pk;
    }
  } else {
    const int nbase = 2048 + h * 64;
    for (int k0 = 0; k0 < DM; k0 += 32) {
      ABf Am;  // A = x rows
      const bf16_t* xr = xb + (size_t)(t0 + l16) * DM + k0 + half * 8;
      Am.u[0] = *(const u32x4*)(xr);
      Am.u[1] = *(const u32x4*)(xr + 16);
#pragma unroll
      for (int dt = 0; dt < 4; ++dt) {
        ABf Bm;  // B = W^T
        const bf16_t* wr = wb + (size_t)(nbase + dt * 16 + l16) * DM + k0 + half * 8;
        Bm.u[0] = *(const u32x4*)(wr);
        Bm.u[1] = *(const u32x4*)(wr + 16);
        acc[dt] = WMMA_BF16(Am.v, Bm.v, acc[dt]);
      }
    }
    const int b  = t0 >> 11;
    const int s0 = (t0 & 2047) + half * 8;
#pragma unroll
    for (int dt = 0; dt < 4; ++dt) {
      u32x4 pk;
#pragma unroll
      for (int jj = 0; jj < 4; ++jj)
        pk[jj] = pack2(acc[dt][2 * jj], acc[dt][2 * jj + 1]);
      *(u32x4*)(VTo + ((size_t)(b * HH + h) * DH + dt * 16 + l16) * SS + s0) = pk;
    }
  }
}

// ---------------------------------------------------------------------------
// Kernel 2: flash attention with LDS-staged K/V^T shared by 8 waves.
//   Block = 8 waves = 128 queries of one (b,h). 64-key chunks are async-copied
//   (double-buffered) into LDS; each wave runs S^T = K*Q^T -> online softmax
//   -> O^T += V^T * P^T, all matrix math via WMMA.
// ---------------------------------------------------------------------------
__global__ __launch_bounds__(256) void fa_kernel(
    const bf16_t* __restrict__ Qi, const bf16_t* __restrict__ Ki,
    const bf16_t* __restrict__ VTi,
    const float* __restrict__ bias_p, const float* __restrict__ bias_a,
    bf16_t* __restrict__ AO) {        // [B*S, 1024]
  __shared__ bf16_t Kbuf[2][64 * LPAD];
  __shared__ bf16_t Vbuf[2][64 * LPAD];

  const int tid  = threadIdx.x;
  const int lane = tid & 31;
  const int wave = tid >> 5;
  const int blk  = blockIdx.x;
  const int qb = blk & 15;             // 16 query-blocks of 128
  const int h  = (blk >> 4) & 15;
  const int b  = blk >> 8;
  const int half = lane >> 4;
  const int l16  = lane & 15;

  const size_t bh = (size_t)(b * HH + h);
  const bf16_t* Qb = Qi  + bh * SS * DH;
  const bf16_t* Kb = Ki  + bh * SS * DH;
  const bf16_t* Vb = VTi + bh * DH * SS;
  const float p = fmaxf(bias_p[h], 0.01f);
  const float a = fmaxf(bias_a[h], 0.01f);
  const int   q0   = (qb * 8 + wave) * 16;
  const float qpos = (float)(q0 + l16);

  // Q^T B-operands (2 d-chunks of 32), loaded once from global
  ABf Bq[2];
#pragma unroll
  for (int dc = 0; dc < 2; ++dc) {
    const bf16_t* ptr = Qb + (size_t)(q0 + l16) * DH + dc * 32 + half * 8;
    Bq[dc].u[0] = *(const u32x4*)(ptr);
    Bq[dc].u[1] = *(const u32x4*)(ptr + 16);
  }

  v8f acc[4];
#pragma unroll
  for (int i = 0; i < 4; ++i) acc[i] = vzero8();
  float m = -1e30f, lsum = 0.0f;

  // ---- async stage of one 64-key chunk: 512 x 16B segments for K and V^T ----
  auto stage = [&](int kc, int buf) {
#pragma unroll
    for (int r = 0; r < 2; ++r) {
      const int seg = tid + r * 256;       // 0..511
      const int row = seg >> 3;            // 64 rows
      const int col = (seg & 7) * 8;       // elem offset in row
      async_load_b128(Kb + (size_t)(kc + row) * DH + col, &Kbuf[buf][row * LPAD + col]);
      async_load_b128(Vb + (size_t)row * SS + kc + col,   &Vbuf[buf][row * LPAD + col]);
    }
  };

  stage(0, 0);
  wait_async0();
  __syncthreads();

  int cur = 0;
  for (int kc = 0; kc < SS; kc += 64) {
    if (kc + 64 < SS) stage(kc + 64, cur ^ 1);
    const bf16_t* kl = Kbuf[cur];
    const bf16_t* vl = Vbuf[cur];

#pragma unroll
    for (int sub = 0; sub < 2; ++sub) {
      const int kb0 = kc + sub * 32;
      // ---- scores^T: two 16-key tiles from LDS ----
      v8f st[2];
      st[0] = vzero8(); st[1] = vzero8();
#pragma unroll
      for (int kt = 0; kt < 2; ++kt) {
        const bf16_t* kr = kl + (sub * 32 + kt * 16 + l16) * LPAD;
#pragma unroll
        for (int dc = 0; dc < 2; ++dc) {
          ABf Am;
          const bf16_t* ptr = kr + dc * 32 + half * 8;
          Am.u[0] = *(const u32x4*)(ptr);
          Am.u[1] = *(const u32x4*)(ptr + 16);
          st[kt] = WMMA_BF16(Am.v, Bq[dc].v, st[kt]);
        }
      }
      // ---- log-distance bias + chunk max ----
      float sv[16];
      float cmax = -1e30f;
#pragma unroll
      for (int kt = 0; kt < 2; ++kt)
#pragma unroll
        for (int j = 0; j < 8; ++j) {
          const float kpos = (float)(kb0 + kt * 16 + half * 8 + j);
          const float dist = fabsf(qpos - kpos);
          const float v = st[kt][j] - p * __logf(fmaf(a, dist, 1.0f));
          sv[kt * 8 + j] = v;
          cmax = fmaxf(cmax, v);
        }
      cmax = fmaxf(cmax, __shfl_xor(cmax, 16, 32));
      const float mnew = fmaxf(m, cmax);
      const float corr = __expf(m - mnew);
      m = mnew;
      float pv[16];
      float csum = 0.0f;
#pragma unroll
      for (int i = 0; i < 16; ++i) { pv[i] = __expf(sv[i] - mnew); csum += pv[i]; }
      csum += __shfl_xor(csum, 16, 32);
      lsum = lsum * corr + csum;
#pragma unroll
      for (int mt = 0; mt < 4; ++mt)
#pragma unroll
        for (int j = 0; j < 8; ++j) acc[mt][j] *= corr;

      // ---- build P^T B-operand via lane-half exchange ----
      ABf Bp;
#pragma unroll
      for (int jj = 0; jj < 4; ++jj) {
        const float a0 = pv[2 * jj],     a1 = pv[2 * jj + 1];      // tile0 pair
        const float b0 = pv[8 + 2 * jj], b1 = pv[8 + 2 * jj + 1];  // tile1 pair
        const float x0 = __shfl_xor(a0, 16, 32), x1 = __shfl_xor(a1, 16, 32);
        const float y0 = __shfl_xor(b0, 16, 32), y1 = __shfl_xor(b1, 16, 32);
        const float lo0 = half ? y0 : a0, lo1 = half ? y1 : a1;
        const float hi0 = half ? b0 : x0, hi1 = half ? b1 : x1;
        Bp.u[0][jj] = pack2(lo0, lo1);
        Bp.u[1][jj] = pack2(hi0, hi1);
      }
      // ---- O^T += V^T * P^T over 4 d-tiles (A from LDS) ----
#pragma unroll
      for (int mt = 0; mt < 4; ++mt) {
        ABf Av;
        const bf16_t* vr = vl + (mt * 16 + l16) * LPAD + sub * 32;
        Av.u[0] = *(const u32x4*)(vr + half * 16);
        Av.u[1] = *(const u32x4*)(vr + half * 16 + 8);
        acc[mt] = WMMA_BF16(Av.v, Bp.v, acc[mt]);
      }
    }
    wait_async0();      // next chunk landed
    __syncthreads();    // everyone done reading `cur` before it is overwritten
    cur ^= 1;
  }

  const float inv = 1.0f / lsum;
  bf16_t* dst = AO + ((size_t)b * SS + q0 + l16) * DM + h * DH;
#pragma unroll
  for (int mt = 0; mt < 4; ++mt) {
    u32x4 pk;
#pragma unroll
    for (int jj = 0; jj < 4; ++jj)
      pk[jj] = pack2(acc[mt][2 * jj] * inv, acc[mt][2 * jj + 1] * inv);
    *(u32x4*)(dst + mt * 16 + half * 8) = pk;
  }
}

// ---------------------------------------------------------------------------
// Kernel 3: output projection out = AO @ out_w^T (f32 result)
// ---------------------------------------------------------------------------
__global__ __launch_bounds__(256) void outproj_kernel(
    const bf16_t* __restrict__ AO,   // [4096,1024] bf16
    const bf16_t* __restrict__ Wo,   // [1024,1024] bf16
    float* __restrict__ out) {       // [4096,1024] f32
  const int lane = threadIdx.x & 31;
  const int wid  = blockIdx.x * 8 + (threadIdx.x >> 5);
  const int mt = wid & 255;
  const int ns = wid >> 8;
  const int half = lane >> 4;
  const int l16  = lane & 15;
  const int t0 = mt * 16;
  const int n0 = ns * 64;

  v8f acc[4];
#pragma unroll
  for (int i = 0; i < 4; ++i) acc[i] = vzero8();

  for (int k0 = 0; k0 < DM; k0 += 32) {
    ABf Bm;  // B = AO^T : lane = token
    const bf16_t* ar = AO + (size_t)(t0 + l16) * DM + k0 + half * 8;
    Bm.u[0] = *(const u32x4*)(ar);
    Bm.u[1] = *(const u32x4*)(ar + 16);
#pragma unroll
    for (int dt = 0; dt < 4; ++dt) {
      ABf Am;  // A = Wo rows
      const bf16_t* wr = Wo + (size_t)(n0 + dt * 16 + l16) * DM + k0 + half * 8;
      Am.u[0] = *(const u32x4*)(wr);
      Am.u[1] = *(const u32x4*)(wr + 16);
      acc[dt] = WMMA_BF16(Am.v, Bm.v, acc[dt]);
    }
  }
  float* dst = out + (size_t)(t0 + l16) * DM + n0;
#pragma unroll
  for (int dt = 0; dt < 4; ++dt) {
    f32x4 s0, s1;
#pragma unroll
    for (int jj = 0; jj < 4; ++jj) { s0[jj] = acc[dt][jj]; s1[jj] = acc[dt][4 + jj]; }
    *(f32x4*)(dst + dt * 16 + half * 8)     = s0;
    *(f32x4*)(dst + dt * 16 + half * 8 + 4) = s1;
  }
}

// ---------------------------------------------------------------------------
// Host-side launcher
// ---------------------------------------------------------------------------
extern "C" void kernel_launch(void* const* d_in, const int* in_sizes, int n_in,
                              void* d_out, int out_size, void* d_ws, size_t ws_size,
                              hipStream_t stream) {
  (void)in_sizes; (void)n_in; (void)out_size; (void)ws_size;
  const float* x      = (const float*)d_in[0];
  const float* qkv_w  = (const float*)d_in[1];
  const float* out_w  = (const float*)d_in[2];
  const float* bias_p = (const float*)d_in[3];
  const float* bias_a = (const float*)d_in[4];
  const float* rope   = (const float*)d_in[5];

  char* ws = (char*)d_ws;
  size_t off = 0;
  auto alloc = [&](size_t bytes) {
    void* pp = ws + off;
    off += (bytes + 255) & ~(size_t)255;
    return pp;
  };
  bf16_t* xb   = (bf16_t*)alloc((size_t)4096 * 1024 * 2);
  bf16_t* wqkv = (bf16_t*)alloc((size_t)3072 * 1024 * 2);
  bf16_t* wo   = (bf16_t*)alloc((size_t)1024 * 1024 * 2);
  bf16_t* Qp   = (bf16_t*)alloc((size_t)BB * HH * SS * DH * 2);
  bf16_t* Kp   = (bf16_t*)alloc((size_t)BB * HH * SS * DH * 2);
  bf16_t* VTp  = (bf16_t*)alloc((size_t)BB * HH * DH * SS * 2);
  bf16_t* AO   = (bf16_t*)alloc((size_t)4096 * 1024 * 2);

  cvt_f32_to_bf16<<<1024, 256, 0, stream>>>(x,     (unsigned short*)xb,   4096 * 1024);
  cvt_f32_to_bf16<<<1024, 256, 0, stream>>>(qkv_w, (unsigned short*)wqkv, 3072 * 1024);
  cvt_f32_to_bf16<<<512,  256, 0, stream>>>(out_w, (unsigned short*)wo,   1024 * 1024);

  qkv_rope_kernel<<<1536, 256, 0, stream>>>(xb, wqkv, rope, Qp, Kp, VTp);

  // B*H*(S/128) = 512 blocks, each: 8 waves share one (b,h), LDS-staged K/V
  fa_kernel<<<512, 256, 0, stream>>>(Qp, Kp, VTp, bias_p, bias_a, AO);

  outproj_kernel<<<512, 256, 0, stream>>>(AO, wo, (float*)d_out);
}
